// DMPNNEncoder_head_9861244912344
// MI455X (gfx1250) — matrix operand order repaired and checked by reading, not compiled, not run on hardware
//
#include <hip/hip_runtime.h>
#include <hip/hip_bf16.h>

// ---------------------------------------------------------------------------
// DMPNN encoder head, fused for MI455X (gfx1250, wave32, WMMA).
//  - edge update fused: m = agg[src] - h[e^1];  h = relu(h0 + m @ W2)
//    via v_wmma_f32_16x16x32_bf16 (bf16 in, f32 accumulate).
//  - reverse edge is e^1 by construction; 16-aligned tiles contain both
//    directions -> in-place second iteration.
//  - segment sums done atomics-free via a per-launch CSR (deg/scan/perm,
//    segments canonically sorted -> deterministic f32 summation order).
// ---------------------------------------------------------------------------

typedef __attribute__((ext_vector_type(16))) __bf16 v16bf;
typedef __attribute__((ext_vector_type(8)))  float  v8f;

#define HID 128
#define GWAVES 4   // waves per block in the WMMA kernels

union FragBF { uint4 u[2]; v16bf v; };

__device__ __forceinline__ unsigned short f2bf(float f) {
  unsigned u = __float_as_uint(f);
  u += 0x7FFFu + ((u >> 16) & 1u);          // round-to-nearest-even
  return (unsigned short)(u >> 16);
}
__device__ __forceinline__ unsigned pk2bf(float lo, float hi) {
  unsigned ul = __float_as_uint(lo), uh = __float_as_uint(hi);
  ul += 0x7FFFu + ((ul >> 16) & 1u);
  uh += 0x7FFFu + ((uh >> 16) & 1u);
  return (ul >> 16) | (uh & 0xFFFF0000u);
}

// -------------------------- small utility kernels --------------------------

__global__ void k_zero_i32(int* __restrict__ p, int n) {
  int i = blockIdx.x * blockDim.x + threadIdx.x;
  if (i < n) p[i] = 0;
}

// Wt[n*K + k] = bf16( W[k*Ncol + n] )
__global__ void k_w_transpose_bf16(const float* __restrict__ W, unsigned short* __restrict__ Wt,
                                   int K, int Ncol) {
  int t = blockIdx.x * blockDim.x + threadIdx.x;
  if (t >= K * Ncol) return;
  int n = t / K, k = t - n * K;
  Wt[t] = f2bf(W[k * Ncol + n]);
}

// -------------------------- CSR build (once per launch) --------------------

__global__ void k_hist(const int* __restrict__ idx, int* __restrict__ deg, long n) {
  long e = (long)blockIdx.x * blockDim.x + threadIdx.x;
  if (e < n) atomicAdd(&deg[idx[e]], 1);
}

// per-256-chunk inclusive scan + block totals
__global__ void k_scan_block(const int* __restrict__ deg, int* __restrict__ incl,
                             int* __restrict__ bsum, int n) {
  __shared__ int s[256];
  int i = blockIdx.x * 256 + threadIdx.x;
  int v = (i < n) ? deg[i] : 0;
  s[threadIdx.x] = v;
  __syncthreads();
  for (int d = 1; d < 256; d <<= 1) {
    int t = (threadIdx.x >= d) ? s[threadIdx.x - d] : 0;
    __syncthreads();
    s[threadIdx.x] += t;
    __syncthreads();
  }
  if (i < n) incl[i] = s[threadIdx.x];
  if (threadIdx.x == 255) bsum[blockIdx.x] = s[255];
}

// single-block exclusive scan (nb <= 1024)
__global__ void k_scan_single(const int* __restrict__ in, int* __restrict__ excl, int nb) {
  __shared__ int s[1024];
  int v = (threadIdx.x < nb) ? in[threadIdx.x] : 0;
  s[threadIdx.x] = v;
  __syncthreads();
  for (int d = 1; d < 1024; d <<= 1) {
    int t = (threadIdx.x >= d) ? s[threadIdx.x - d] : 0;
    __syncthreads();
    s[threadIdx.x] += t;
    __syncthreads();
  }
  if (threadIdx.x < nb) excl[threadIdx.x] = s[threadIdx.x] - v;
}

__global__ void k_finalize_off(const int* __restrict__ incl, const int* __restrict__ deg,
                               const int* __restrict__ boff, int* __restrict__ offsets,
                               int* __restrict__ cursor, int n) {
  int i = blockIdx.x * blockDim.x + threadIdx.x;
  if (i >= n) return;
  int ex = incl[i] - deg[i] + boff[i >> 8];
  offsets[i] = ex;
  cursor[i]  = ex;
}

__global__ void k_fill_perm(const int* __restrict__ idx, int* __restrict__ cursor,
                            int* __restrict__ perm, long n) {
  long e = (long)blockIdx.x * blockDim.x + threadIdx.x;
  if (e < n) {
    int p = atomicAdd(&cursor[idx[e]], 1);
    perm[p] = (int)e;
  }
}

// canonicalize each segment (ascending edge id) -> deterministic sums
__global__ void k_sort_seg(int* __restrict__ perm, const int* __restrict__ offsets,
                           const int* __restrict__ deg, int nseg) {
  int n = blockIdx.x * blockDim.x + threadIdx.x;
  if (n >= nseg) return;
  int s = offsets[n], d = deg[n];
  for (int i = 1; i < d; ++i) {
    int key = perm[s + i];
    int j = i - 1;
    while (j >= 0 && perm[s + j] > key) { perm[s + j + 1] = perm[s + j]; --j; }
    perm[s + j + 1] = key;
  }
}

// -------------------------- atomics-free segment sums ----------------------
// one wave per segment; lanes own 4 columns; coalesced 512B row gathers

__global__ __launch_bounds__(256)
void k_segsum(const float* __restrict__ vals, const int* __restrict__ perm,
              const int* __restrict__ offsets, const int* __restrict__ deg,
              float* __restrict__ out, int nseg) {
  int wave = threadIdx.x >> 5, lane = threadIdx.x & 31;
  int n = blockIdx.x * 8 + wave;
  if (n >= nseg) return;
  int start = offsets[n], d = deg[n];
  float4 acc = make_float4(0.f, 0.f, 0.f, 0.f);
  for (int j = 0; j < d; ++j) {
    int e = perm[start + j];
    const float4 v = *(const float4*)(vals + (long)e * HID + (lane << 2));
    acc.x += v.x; acc.y += v.y; acc.z += v.z; acc.w += v.w;
  }
  *(float4*)(out + (long)n * HID + (lane << 2)) = acc;
}

// contiguous-range variant (graph pooling; batch is sorted)
__global__ __launch_bounds__(256)
void k_segsum_range(const float* __restrict__ vals, const int* __restrict__ offsets,
                    const int* __restrict__ deg, float* __restrict__ out, int nseg) {
  int wave = threadIdx.x >> 5, lane = threadIdx.x & 31;
  int n = blockIdx.x * 8 + wave;
  if (n >= nseg) return;
  int start = offsets[n], d = deg[n];
  float4 acc = make_float4(0.f, 0.f, 0.f, 0.f);
  for (int j = 0; j < d; ++j) {
    const float4 v = *(const float4*)(vals + (long)(start + j) * HID + (lane << 2));
    acc.x += v.x; acc.y += v.y; acc.z += v.z; acc.w += v.w;
  }
  *(float4*)(out + (long)n * HID + (lane << 2)) = acc;
}

// -------------------------- fused edge update ------------------------------
// One wave per 16-edge tile:  h_out[e] = relu(h0[e] + (agg[src[e]] - h_old[e^1]) @ W2)

__global__ __launch_bounds__(GWAVES * 32)
void k_edge_update(const float* h_old, const float* __restrict__ h0,
                   const float* __restrict__ agg, const int* __restrict__ src,
                   const unsigned short* __restrict__ w2t,   // bf16 [n][k] (n-major)
                   float* h_out, int ntiles) {
  __shared__ unsigned short w2s[HID * HID];          // 32 KB
  __shared__ unsigned short ms[GWAVES][16 * HID];    // 16 KB (bf16 m-tiles)

  {
    const uint4* gsrc = (const uint4*)w2t;
    uint4* gdst = (uint4*)w2s;
    for (int i = threadIdx.x; i < (HID * HID) / 8; i += blockDim.x) gdst[i] = gsrc[i];
  }

  const int wave = threadIdx.x >> 5, lane = threadIdx.x & 31;
  const int tile = blockIdx.x * GWAVES + wave;
  const bool active = tile < ntiles;

  if (active) {
    int r  = lane >> 1;
    int ch = (lane & 1) << 6;
    int e  = tile * 16 + r;
    const float4* ap = (const float4*)(agg  + (long)src[e] * HID + ch);
    const float4* hp = (const float4*)(h_old + (long)(e ^ 1) * HID + ch);
    uint2* st = (uint2*)&ms[wave][r * HID + ch];
#pragma unroll
    for (int i = 0; i < 16; ++i) {
      float4 a4 = ap[i], r4 = hp[i];
      uint2 p;
      p.x = pk2bf(a4.x - r4.x, a4.y - r4.y);
      p.y = pk2bf(a4.z - r4.z, a4.w - r4.w);
      st[i] = p;
    }
  }
  __syncthreads();
  if (!active) return;

  const int M = lane & 15, hi = lane >> 4;

  FragBF A[4];
  const unsigned short* mrow = &ms[wave][M * HID];
#pragma unroll
  for (int kb = 0; kb < 4; ++kb) {
    int k0 = kb * 32 + hi * 16;
    A[kb].u[0] = *(const uint4*)(mrow + k0);
    A[kb].u[1] = *(const uint4*)(mrow + k0 + 8);
  }

#pragma unroll
  for (int nt = 0; nt < 8; ++nt) {
    v8f C = {};
#pragma unroll
    for (int kb = 0; kb < 4; ++kb) {
      FragBF B;
      const unsigned short* brow = &w2s[(nt * 16 + M) * HID + kb * 32 + hi * 16];
      B.u[0] = *(const uint4*)(brow);
      B.u[1] = *(const uint4*)(brow + 8);
      C = __builtin_amdgcn_wmma_f32_16x16x32_bf16(false, A[kb].v, false, B.v,
                                                  (short)0, C, false, false);
    }
    const int n = nt * 16 + M;
    const int rb = hi * 8;
#pragma unroll
    for (int v = 0; v < 8; ++v) {
      long idx = (long)(tile * 16 + rb + v) * HID + n;
      float d = C[v] + h0[idx];
      h_out[idx] = d > 0.f ? d : 0.f;
    }
  }
}

// -------------------------- fused node update ------------------------------

__global__ __launch_bounds__(GWAVES * 32)
void k_node_update(const float* __restrict__ x, const float* __restrict__ vmsg,
                   const unsigned short* __restrict__ w3t,   // bf16 [n][k], k-stride 256
                   float* __restrict__ node_attr, int nnodes) {
  __shared__ unsigned short zs[GWAVES][16 * 256];   // 32 KB
  const int wave = threadIdx.x >> 5, lane = threadIdx.x & 31;
  const int ntiles = (nnodes + 15) >> 4;
  const int tile = blockIdx.x * GWAVES + wave;
  const bool active = tile < ntiles;

  if (active) {
    int r    = lane >> 1;
    int half = lane & 1;
    int node = tile * 16 + r;
    int nc   = node < nnodes ? node : 0;
    const float4* p = (const float4*)((half ? vmsg : x) + (long)nc * HID);
    uint2* st = (uint2*)&zs[wave][r * 256 + half * HID];
#pragma unroll
    for (int i = 0; i < 32; ++i) {
      float4 v4 = p[i];
      uint2 q;
      q.x = pk2bf(v4.x, v4.y);
      q.y = pk2bf(v4.z, v4.w);
      st[i] = q;
    }
  }
  __syncthreads();
  if (!active) return;

  const int M = lane & 15, hi = lane >> 4;
  FragBF A[8];
  const unsigned short* zrow = &zs[wave][M * 256];
#pragma unroll
  for (int kb = 0; kb < 8; ++kb) {
    int k0 = kb * 32 + hi * 16;
    A[kb].u[0] = *(const uint4*)(zrow + k0);
    A[kb].u[1] = *(const uint4*)(zrow + k0 + 8);
  }

#pragma unroll
  for (int nt = 0; nt < 8; ++nt) {
    v8f C = {};
#pragma unroll
    for (int kb = 0; kb < 8; ++kb) {
      FragBF B;
      const unsigned short* brow = w3t + (long)(nt * 16 + M) * 256 + kb * 32 + hi * 16;
      B.u[0] = *(const uint4*)(brow);
      B.u[1] = *(const uint4*)(brow + 8);
      C = __builtin_amdgcn_wmma_f32_16x16x32_bf16(false, A[kb].v, false, B.v,
                                                  (short)0, C, false, false);
    }
    const int n = nt * 16 + M;
    const int rb = hi * 8;
#pragma unroll
    for (int v = 0; v < 8; ++v) {
      int node = tile * 16 + rb + v;
      if (node < nnodes) {
        float d = C[v];
        node_attr[(long)node * HID + n] = d > 0.f ? d : 0.f;
      }
    }
  }
}

// -------------------------- tiny graph head --------------------------------

__global__ void k_head1(const float* __restrict__ g, const float* __restrict__ W,
                        const float* __restrict__ b, float* __restrict__ t1, int G) {
  int t = blockIdx.x * blockDim.x + threadIdx.x;
  if (t >= G * HID) return;
  int gi = t >> 7, j = t & (HID - 1);
  float acc = b[j];
  const float* gr = g + gi * HID;
#pragma unroll 4
  for (int k = 0; k < HID; ++k) acc = fmaf(gr[k], W[k * HID + j], acc);
  t1[t] = acc > 0.f ? acc : 0.f;
}

__global__ void k_head2(const float* __restrict__ t1, const float* __restrict__ W,
                        const float* __restrict__ b, float* __restrict__ out, int G) {
  int t = blockIdx.x * blockDim.x + threadIdx.x;
  if (t >= G * 64) return;
  int gi = t >> 6, j = t & 63;
  float acc = b[j];
  const float* tr = t1 + gi * HID;
#pragma unroll 4
  for (int k = 0; k < HID; ++k) acc = fmaf(tr[k], W[k * 64 + j], acc);
  out[t] = acc;
}

// ---------------------------------------------------------------------------

extern "C" void kernel_launch(void* const* d_in, const int* in_sizes, int n_in,
                              void* d_out, int out_size, void* d_ws, size_t ws_size,
                              hipStream_t stream) {
  (void)n_in; (void)out_size; (void)ws_size;
  const float* x   = (const float*)d_in[0];
  const int*   ei  = (const int*)d_in[1];
  const float* ea  = (const float*)d_in[2];   // edge_attr == h0
  const int* batch = (const int*)d_in[3];
  const float* W2  = (const float*)d_in[4];
  const float* W3  = (const float*)d_in[5];
  const float* HW1 = (const float*)d_in[6];
  const float* Hb1 = (const float*)d_in[7];
  const float* HW2 = (const float*)d_in[8];
  const float* Hb2 = (const float*)d_in[9];
  float* out = (float*)d_out;

  const long E = in_sizes[1] / 2;
  const int  N = in_sizes[3];
  const int  G = 64, OUT = 64;
  const int* src = ei;
  const int* dst = ei + E;

  // workspace carve-up (floats first -> 16B alignment for vector loads)
  char* ws = (char*)d_ws;
  float* h = (float*)ws;                          ws += (size_t)E * HID * 4;
  float* agg = (float*)ws;                        ws += (size_t)N * HID * 4;
  float* node_attr = (float*)ws;                  ws += (size_t)N * HID * 4;
  unsigned short* w2t = (unsigned short*)ws;      ws += (size_t)HID * HID * 2;
  unsigned short* w3t = (unsigned short*)ws;      ws += (size_t)HID * 256 * 2;
  float* g  = (float*)ws;                         ws += (size_t)G * HID * 4;
  float* t1 = (float*)ws;                         ws += (size_t)G * HID * 4;
  int* deg     = (int*)ws;                        ws += (size_t)N * 4;
  int* incl    = (int*)ws;                        ws += (size_t)N * 4;
  int* offs    = (int*)ws;                        ws += (size_t)N * 4;
  int* cursor  = (int*)ws;                        ws += (size_t)N * 4;
  int* bsum    = (int*)ws;                        ws += (size_t)1024 * 4;
  int* boff    = (int*)ws;                        ws += (size_t)1024 * 4;
  int* gdeg    = (int*)ws;                        ws += (size_t)64 * 4;
  int* goff    = (int*)ws;                        ws += (size_t)64 * 4;
  int* perm    = (int*)ws;

  // one-time weight transpose + bf16 convert
  k_w_transpose_bf16<<<(HID * HID + 255) / 256, 256, 0, stream>>>(W2, w2t, HID, HID);
  k_w_transpose_bf16<<<(256 * HID + 255) / 256, 256, 0, stream>>>(W3, w3t, 256, HID);

  // ---- build CSR over dst (used by all three segment sums) ----
  const int nb = (N + 255) / 256;                 // <= 1024 assumed
  const int eb = (int)((E + 255) / 256);
  k_zero_i32<<<(N + 255) / 256, 256, 0, stream>>>(deg, N);
  k_hist<<<eb, 256, 0, stream>>>(dst, deg, E);
  k_scan_block<<<nb, 256, 0, stream>>>(deg, incl, bsum, N);
  k_scan_single<<<1, 1024, 0, stream>>>(bsum, boff, nb);
  k_finalize_off<<<(N + 255) / 256, 256, 0, stream>>>(incl, deg, boff, offs, cursor, N);
  k_fill_perm<<<eb, 256, 0, stream>>>(dst, cursor, perm, E);
  k_sort_seg<<<(N + 255) / 256, 256, 0, stream>>>(perm, offs, deg, N);

  const int segb = (N + 7) / 8;
  const int etiles = (int)(E / 16);
  const int eblocks = (etiles + GWAVES - 1) / GWAVES;

  // ---- MP iteration 1 (h starts as h0) ----
  k_segsum<<<segb, 256, 0, stream>>>(ea, perm, offs, deg, agg, N);
  k_edge_update<<<eblocks, GWAVES * 32, 0, stream>>>(ea, ea, agg, src, w2t, h, etiles);

  // ---- MP iteration 2 (in place: pairs e/e^1 share a 16-tile) ----
  k_segsum<<<segb, 256, 0, stream>>>(h, perm, offs, deg, agg, N);
  k_edge_update<<<eblocks, GWAVES * 32, 0, stream>>>(h, ea, agg, src, w2t, h, etiles);

  // ---- v_msg = segment_sum(h, dst) ----
  k_segsum<<<segb, 256, 0, stream>>>(h, perm, offs, deg, agg, N);

  // ---- node_attr = relu([x | v_msg] @ W3) ----
  const int ntiles = (N + 15) / 16;
  const int nblocks = (ntiles + GWAVES - 1) / GWAVES;
  k_node_update<<<nblocks, GWAVES * 32, 0, stream>>>(x, agg, w3t, node_attr, N);

  // ---- graph pooling (batch sorted -> contiguous ranges) ----
  k_zero_i32<<<1, 64, 0, stream>>>(gdeg, G);
  k_hist<<<(N + 255) / 256, 256, 0, stream>>>(batch, gdeg, N);
  k_scan_single<<<1, 1024, 0, stream>>>(gdeg, goff, G);
  k_segsum_range<<<(G + 7) / 8, 256, 0, stream>>>(node_attr, goff, gdeg, g, G);

  // ---- head ----
  k_head1<<<(G * HID + 255) / 256, 256, 0, stream>>>(g, HW1, Hb1, t1, G);
  k_head2<<<(G * OUT + 255) / 256, 256, 0, stream>>>(t1, HW2, Hb2, out, G);
}
